// DecoderNoTrainer_67559835566086
// MI455X (gfx1250) — matrix operand order repaired and checked by reading, not compiled, not run on hardware
//
#include <hip/hip_runtime.h>

typedef __attribute__((ext_vector_type(16))) __bf16 bf16x16;
typedef __attribute__((ext_vector_type(8)))  __bf16 bf16x8;
typedef __attribute__((ext_vector_type(8)))  float  f32x8;

#define AW  520   // padded LDS A width (bf16 elems): 1040B row stride -> no bank conflicts
#define EPW 264   // padded LDS A width for encproj (K=256)

__device__ __forceinline__ float fast_rcp(float x) { return __builtin_amdgcn_rcpf(x); }
__device__ __forceinline__ float tanh_fast(float x) {
    float e = __expf(2.0f * x);
    return 1.0f - 2.0f * fast_rcp(e + 1.0f);
}
__device__ __forceinline__ float sigmoid_fast(float x) {
    return fast_rcp(1.0f + __expf(-x));
}

__device__ __forceinline__ f32x8 wmma_bf16(bf16x16 a, bf16x16 b, f32x8 c) {
    // D = A(16x32 bf16) * B(32x16 bf16) + C(16x16 f32)
    return __builtin_amdgcn_wmma_f32_16x16x32_bf16(false, a, false, b, (short)0, c, false, false);
}

// A fragment from LDS row-major [16 x width] bf16; chunk kc covers K=[kc*32, kc*32+32).
// A layout (matches packed B): lane<16 holds row=lane, K {0..15}; lane>=16 holds K {16..31}.
__device__ __forceinline__ bf16x16 load_a_chunk(const __bf16* Ab, int width, int kc, int lane) {
    int row  = lane & 15;
    int koff = kc * 32 + ((lane < 16) ? 0 : 16);
    const __bf16* p = Ab + row * width + koff;
    bf16x8 lo = *(const bf16x8*)(p);
    bf16x8 hi = *(const bf16x8*)(p + 8);
    bf16x16 r;
#pragma unroll
    for (int i = 0; i < 8; ++i) { r[i] = lo[i]; r[8 + i] = hi[i]; }
    return r;
}

// B fragment, pre-packed in global: tile = nt*(K/32)+kc; 512 bf16 per tile, 16 contiguous per lane.
__device__ __forceinline__ bf16x16 load_b_chunk(const __bf16* Wp, int tile, int lane) {
    const __bf16* p = Wp + ((size_t)tile << 9) + (lane << 4);
    bf16x8 lo = *(const bf16x8*)(p);
    bf16x8 hi = *(const bf16x8*)(p + 8);
    bf16x16 r;
#pragma unroll
    for (int i = 0; i < 8; ++i) { r[i] = lo[i]; r[8 + i] = hi[i]; }
    return r;
}

// ---------- one-time packing kernels ----------

// Pack W[K x N] f32 -> bf16 in WMMA B layout.
// B layout (this code's contiguous K order, same as A): lane n<16: col=n, K = kc*32 + e (e=0..15);
//                                 lane>=16: col=n-16, K = kc*32 + 16 + e.
__global__ void pack_b16(const float* __restrict__ src, __bf16* __restrict__ dst, int K, int N) {
    int idx = blockIdx.x * blockDim.x + threadIdx.x;
    if (idx >= K * N) return;
    int within = idx & 511;
    int blkI   = idx >> 9;
    int lane   = within >> 4;
    int e      = within & 15;
    int kchunks = K >> 5;
    int nt = blkI / kchunks;
    int kc = blkI - nt * kchunks;
    int n  = nt * 16 + (lane & 15);
    int k  = kc * 32 + e + ((lane >= 16) ? 16 : 0);
    dst[idx] = (__bf16)src[k * N + n];
}

// Pack combined [Wk;Wr] (512 x 1024) -> bf16 WMMA B layout.
__global__ void pack_wc(const float* __restrict__ Wk, const float* __restrict__ Wr,
                        __bf16* __restrict__ dst) {
    int idx = blockIdx.x * blockDim.x + threadIdx.x;
    if (idx >= 512 * 1024) return;
    int within = idx & 511;
    int blkI   = idx >> 9;
    int lane   = within >> 4;
    int e      = within & 15;
    int nt = blkI >> 4;   // kchunks = 512/32 = 16
    int kc = blkI & 15;
    int n  = nt * 16 + (lane & 15);
    int k  = kc * 32 + e + ((lane >= 16) ? 16 : 0);
    const float* s = (k < 256) ? Wk : Wr;
    dst[idx] = (__bf16)s[(k & 255) * 1024 + n];
}

__global__ void conv_bf16(const float* __restrict__ src, __bf16* __restrict__ dst, int n) {
    int i = blockIdx.x * blockDim.x + threadIdx.x;
    if (i < n) dst[i] = (__bf16)src[i];
}

// ---------- enc_proj = enc_outputs @ W1 + b1 (hoisted) ----------
__global__ __launch_bounds__(256)
void encproj_kernel(const float* __restrict__ enc, const __bf16* __restrict__ W1p,
                    const float* __restrict__ b1, float* __restrict__ encp) {
    __shared__ __align__(16) __bf16 Ab[16 * EPW];
    int tid = threadIdx.x, lane = tid & 31, w = tid >> 5;
    size_t row0 = (size_t)blockIdx.x * 16;

#pragma unroll
    for (int it = 0; it < 4; ++it) {
        int idx = tid + it * 256;            // 1024 float4 units = 16x256 floats
        int r = idx >> 6, c4 = (idx & 63) << 2;
        float4 v = *(const float4*)(enc + (row0 + r) * 256 + c4);
        __bf16* d = &Ab[r * EPW + c4];
        d[0] = (__bf16)v.x; d[1] = (__bf16)v.y; d[2] = (__bf16)v.z; d[3] = (__bf16)v.w;
    }
    __syncthreads();

    f32x8 acc0 = {}, acc1 = {};
    int nt0 = w * 2, nt1 = nt0 + 1;
#pragma unroll
    for (int kc = 0; kc < 8; ++kc) {
        bf16x16 a = load_a_chunk(Ab, EPW, kc, lane);
        acc0 = wmma_bf16(a, load_b_chunk(W1p, nt0 * 8 + kc, lane), acc0);
        acc1 = wmma_bf16(a, load_b_chunk(W1p, nt1 * 8 + kc, lane), acc1);
    }
    int n = lane & 15, mb = (lane < 16) ? 0 : 8;
    int col0 = nt0 * 16 + n, col1 = nt1 * 16 + n;
    float bb0 = b1[col0], bb1 = b1[col1];
#pragma unroll
    for (int i = 0; i < 8; ++i) {
        encp[(row0 + mb + i) * 256 + col0] = acc0[i] + bb0;
        encp[(row0 + mb + i) * 256 + col1] = acc1[i] + bb1;
    }
}

// ---------- persistent decoder: 64 blocks x 512 threads, 128 steps ----------
__global__ __launch_bounds__(512)
void decoder_kernel(const float* __restrict__ dec_h, const float* __restrict__ dec_c,
                    const int* __restrict__ enc_input,
                    const __bf16* __restrict__ Ep, const __bf16* __restrict__ Wcp,
                    const __bf16* __restrict__ W2p,
                    const float* __restrict__ bg, const float* __restrict__ b2,
                    const float* __restrict__ V,
                    const float* __restrict__ encp, float* __restrict__ out) {
    __shared__ __align__(16) __bf16 Ab[16 * AW];        // [x | h] bf16, row-major padded
    __shared__ float zbuf[16 * 1024];                   // gate pre-activations
    __shared__ float cbuf[16 * 256];                    // LSTM cell state (fp32)
    __shared__ __align__(16) float decbuf[16 * 256];    // h @ W2 + b2
    __shared__ float sbuf[16 * 128];                    // attention scores
    __shared__ float bbuf[1024];
    __shared__ float b2buf[256];
    __shared__ float Vbuf[256];
    __shared__ int   tokbuf[16];
    __shared__ float invbuf[16];

    const int tid = threadIdx.x, lane = tid & 31, w = tid >> 5;  // 16 waves
    const int b0 = blockIdx.x * 16;

    for (int i = tid; i < 1024; i += 512) bbuf[i] = bg[i];
    if (tid < 256) { b2buf[tid] = b2[tid]; Vbuf[tid] = V[tid]; }
    for (int idx = tid; idx < 16 * 256; idx += 512) {
        int r = idx >> 8, u = idx & 255;
        cbuf[idx] = dec_c[(b0 + r) * 256 + u];
        Ab[r * AW + 256 + u] = (__bf16)dec_h[(b0 + r) * 256 + u];
    }
    if (tid < 16) tokbuf[tid] = 1;  // SOS token
    __syncthreads();

    for (int t = 0; t < 128; ++t) {
        // 1. x = E[tok] -> A[:, 0:256]
        {
            int r   = tid >> 5;          // 0..15
            int seg = (lane) * 8;        // 0..248
            int tok = tokbuf[r];
            bf16x8 v = *(const bf16x8*)(Ep + tok * 256 + seg);
            *(bf16x8*)(&Ab[r * AW + seg]) = v;
        }
        __syncthreads();

        // 2. z = [x;h] @ [Wk;Wr] + b   (16x512 @ 512x1024)
        {
            f32x8 acc[4] = {};
            for (int kc = 0; kc < 16; ++kc) {
                bf16x16 a = load_a_chunk(Ab, AW, kc, lane);
#pragma unroll
                for (int j = 0; j < 4; ++j)
                    acc[j] = wmma_bf16(a, load_b_chunk(Wcp, (w * 4 + j) * 16 + kc, lane), acc[j]);
            }
            int n = lane & 15, mb = (lane < 16) ? 0 : 8;
#pragma unroll
            for (int j = 0; j < 4; ++j) {
                int col = (w * 4 + j) * 16 + n;
                float bias = bbuf[col];
#pragma unroll
                for (int i = 0; i < 8; ++i)
                    zbuf[(mb + i) * 1024 + col] = acc[j][i] + bias;
            }
        }
        __syncthreads();

        // 3. LSTM gates (Keras order i,f,g,o); h -> A[:, 256:512] as bf16
        for (int idx = tid; idx < 4096; idx += 512) {
            int r = idx >> 8, u = idx & 255;
            const float* zr = &zbuf[r * 1024 + u];
            float ig = sigmoid_fast(zr[0]);
            float fg = sigmoid_fast(zr[256]);
            float gg = tanh_fast(zr[512]);
            float og = sigmoid_fast(zr[768]);
            float c  = fg * cbuf[idx] + ig * gg;
            cbuf[idx] = c;
            float hn = og * tanh_fast(c);
            Ab[r * AW + 256 + u] = (__bf16)hn;
        }
        __syncthreads();

        // 4. dec = h @ W2 + b2   (16x256 @ 256x256), one N-tile per wave
        {
            f32x8 d = {};
#pragma unroll
            for (int kc = 0; kc < 8; ++kc) {
                bf16x16 a = load_a_chunk(Ab, AW, kc + 8, lane);
                d = wmma_bf16(a, load_b_chunk(W2p, w * 8 + kc, lane), d);
            }
            int n = lane & 15, mb = (lane < 16) ? 0 : 8;
            int col = w * 16 + n;
            float bias = b2buf[col];
#pragma unroll
            for (int i = 0; i < 8; ++i)
                decbuf[(mb + i) * 256 + col] = d[i] + bias;
        }
        __syncthreads();

        // 5. scores[r,l] = V . tanh(enc_proj[b,l,:] + dec[r,:]); wave w owns row r=w
        {
            int ub = lane * 8;
            float vv[8], dd[8];
#pragma unroll
            for (int q = 0; q < 8; ++q) { vv[q] = Vbuf[ub + q]; dd[q] = decbuf[w * 256 + ub + q]; }
            const float* ebase = encp + (((size_t)(b0 + w) * 128) << 8) + ub;
            for (int l = 0; l < 128; ++l) {
                const float* ep = ebase + ((size_t)l << 8);
                float s = 0.f;
#pragma unroll
                for (int q = 0; q < 8; ++q) s += vv[q] * tanh_fast(ep[q] + dd[q]);
#pragma unroll
                for (int off = 16; off > 0; off >>= 1) s += __shfl_xor(s, off, 32);
                if (lane == 0) sbuf[w * 128 + l] = s;
            }
        }
        __syncthreads();

        // 6. softmax + argmax + next token (first max wins, like jnp.argmax)
        if (tid < 16) {
            int r = tid;
            float mx = -3.0e38f; int am = 0;
            for (int l = 0; l < 128; ++l) {
                float s = sbuf[r * 128 + l];
                if (s > mx) { mx = s; am = l; }
            }
            float sum = 0.f;
            for (int l = 0; l < 128; ++l) {
                float e = __expf(sbuf[r * 128 + l] - mx);
                sbuf[r * 128 + l] = e;
                sum += e;
            }
            invbuf[r] = 1.0f / sum;
            tokbuf[r] = enc_input[(b0 + r) * 128 + am];
        }
        __syncthreads();

        // 7. write pointers[b, t, l]
        for (int idx = tid; idx < 2048; idx += 512) {
            int r = idx >> 7, l = idx & 127;
            out[(((size_t)(b0 + r) * 128) + t) * 128 + l] = sbuf[idx] * invbuf[r];
        }
        __syncthreads();
    }
}

extern "C" void kernel_launch(void* const* d_in, const int* in_sizes, int n_in,
                              void* d_out, int out_size, void* d_ws, size_t ws_size,
                              hipStream_t stream) {
    (void)in_sizes; (void)n_in; (void)out_size; (void)ws_size;
    const float* dec_h       = (const float*)d_in[0];
    const float* dec_c       = (const float*)d_in[1];
    const float* enc_outputs = (const float*)d_in[2];
    const int*   enc_input   = (const int*)d_in[3];
    const float* E           = (const float*)d_in[4];
    const float* Wk          = (const float*)d_in[5];
    const float* Wr          = (const float*)d_in[6];
    const float* bg          = (const float*)d_in[7];
    const float* W1          = (const float*)d_in[8];
    const float* b1          = (const float*)d_in[9];
    const float* W2          = (const float*)d_in[10];
    const float* b2          = (const float*)d_in[11];
    const float* V           = (const float*)d_in[12];
    float* out = (float*)d_out;

    char* ws = (char*)d_ws;
    size_t off = 0;
    float*  encp = (float*)(ws + off);  off += (size_t)1024 * 128 * 256 * sizeof(float);
    __bf16* Wcp  = (__bf16*)(ws + off); off += (size_t)512 * 1024 * sizeof(__bf16);
    __bf16* W1p  = (__bf16*)(ws + off); off += (size_t)256 * 256 * sizeof(__bf16);
    __bf16* W2p  = (__bf16*)(ws + off); off += (size_t)256 * 256 * sizeof(__bf16);
    __bf16* Ep   = (__bf16*)(ws + off); off += (size_t)130 * 256 * sizeof(__bf16);

    pack_wc  <<<(512 * 1024 + 255) / 256, 256, 0, stream>>>(Wk, Wr, Wcp);
    pack_b16 <<<(256 * 256 + 255) / 256, 256, 0, stream>>>(W1, W1p, 256, 256);
    pack_b16 <<<(256 * 256 + 255) / 256, 256, 0, stream>>>(W2, W2p, 256, 256);
    conv_bf16<<<(130 * 256 + 255) / 256, 256, 0, stream>>>(E, Ep, 130 * 256);

    encproj_kernel<<<(1024 * 128) / 16, 256, 0, stream>>>(enc_outputs, W1p, b1, encp);
    decoder_kernel<<<1024 / 16, 512, 0, stream>>>(dec_h, dec_c, enc_input,
                                                  Ep, Wcp, W2p, bg, b2, V, encp, out);
}
